// HVQ_64570538328099
// MI455X (gfx1250) — matrix operand order repaired and checked by reading, not compiled, not run on hardware
//
#include <hip/hip_runtime.h>
#include <hip/hip_bf16.h>
#include <math.h>

typedef __attribute__((ext_vector_type(16))) _Float16 v16h;
typedef __attribute__((ext_vector_type(4)))  _Float16 v4h;
typedef __attribute__((ext_vector_type(8)))  float    v8f;
typedef __attribute__((ext_vector_type(4)))  int      v4i;

#define B_DIM 8
#define N_DIM 4096
#define F_DIM 512
#define H_DIM 8
#define S_DIM 1024
#define D_DIM 64
#define ROWS_PER_BLOCK 128
#define STAGE_COLS 64           // codebook rows staged to LDS per round
#define N_STAGES (S_DIM / STAGE_COLS)

#if __has_builtin(__builtin_amdgcn_global_load_async_to_lds_b128)
#define HVQ_ASYNC 1
#else
#define HVQ_ASYNC 0
#endif

typedef __attribute__((address_space(1))) v4i gv4i;
typedef __attribute__((address_space(3))) v4i lv4i;

__device__ __forceinline__ void hvq_wait_async0() {
#if __has_builtin(__builtin_amdgcn_s_wait_asynccnt)
    __builtin_amdgcn_s_wait_asynccnt(0);
#else
    asm volatile("s_wait_asynccnt 0x0" ::: "memory");
#endif
}

// ---------------------------------------------------------------------------
// Kernel A: L2-normalize codebooks [H*S, 64] f32 -> f16 in workspace.
// ---------------------------------------------------------------------------
__global__ void hvq_cbnorm_kernel(const float* __restrict__ cb,
                                  _Float16* __restrict__ cnh) {
    int r = blockIdx.x * blockDim.x + threadIdx.x;
    if (r >= H_DIM * S_DIM) return;
    const float4* s4 = (const float4*)(cb + (size_t)r * D_DIM);
    float ss = 0.f;
#pragma unroll
    for (int j = 0; j < 16; ++j) {
        float4 v = s4[j];
        ss += v.x * v.x + v.y * v.y + v.z * v.z + v.w * v.w;
    }
    float scale = 1.0f / fmaxf(sqrtf(ss), 1e-12f);
    v4h* dst = (v4h*)(cnh + (size_t)r * D_DIM);
#pragma unroll
    for (int j = 0; j < 16; ++j) {
        float4 v = s4[j];
        v4h h;
        h[0] = (_Float16)(v.x * scale);
        h[1] = (_Float16)(v.y * scale);
        h[2] = (_Float16)(v.z * scale);
        h[3] = (_Float16)(v.w * scale);
        dst[j] = h;
    }
}

// ---------------------------------------------------------------------------
// Kernel B: per (b,h) 128-row tile.
// 256 threads = 8 waves. B tiles staged once per block into double-buffered
// LDS (async global->LDS, ASYNCcnt-tracked); all 8 waves consume them via
// ds_load_b128 in contiguous k = 16*half + s order, matching A.
// ---------------------------------------------------------------------------
__global__ void __launch_bounds__(256, 2)
hvq_main_kernel(const float* __restrict__ x,
                const float* __restrict__ cb,
                const _Float16* __restrict__ cnh,
                float* __restrict__ outp,
                float* __restrict__ idxp,
                float* __restrict__ counts) {
    __shared__ float sq[ROWS_PER_BLOCK * D_DIM];                     // 32 KB
    __shared__ float sscale[ROWS_PER_BLOCK];
    __shared__ __attribute__((aligned(32))) _Float16 sB[2][STAGE_COLS * D_DIM]; // 2 x 8 KB

    const int b  = blockIdx.x >> 3;
    const int h  = blockIdx.x & 7;
    const int n0 = blockIdx.y * ROWS_PER_BLOCK;
    const int t  = threadIdx.x;
    const _Float16* cn_h = cnh + (size_t)h * S_DIM * D_DIM;

    // ---- stage q tile: 128 rows x 64 floats ----
    {
        const float* xb = x + ((size_t)b * N_DIM + n0) * F_DIM + h * D_DIM;
#pragma unroll
        for (int k = 0; k < 8; ++k) {
            int v    = t + k * 256;
            int row  = v >> 4;
            int col4 = v & 15;
            float4 val = *(const float4*)(xb + (size_t)row * F_DIM + col4 * 4);
            *(float4*)(&sq[row * D_DIM + col4 * 4]) = val;
        }
    }
    __syncthreads();

    // ---- row norms: 2 lanes per row ----
    {
        int row  = t >> 1;
        int half = t & 1;
        float ss = 0.f;
#pragma unroll
        for (int j = 0; j < 32; ++j) {
            float v = sq[row * D_DIM + half * 32 + j];
            ss += v * v;
        }
        ss += __shfl_xor(ss, 1, 32);
        if (half == 0) sscale[row] = 1.0f / fmaxf(sqrtf(ss), 1e-12f);
    }
    __syncthreads();

    // ---- build A fragments (contiguous k = 16*half + s, matching B), VGPR-resident ----
    const int w  = t >> 5;
    const int L  = t & 31;
    const int m  = L & 15;
    const int kb = (L >> 4) * 16;
    const int lrow = w * 16 + m;
    const float qs = sscale[lrow];

    v16h a0{}, a1{};
#pragma unroll
    for (int i = 0; i < 8; ++i) {
        a0[i]     = (_Float16)(sq[lrow * D_DIM + kb + i]           * qs);
        a0[i + 8] = (_Float16)(sq[lrow * D_DIM + kb + 8 + i]       * qs);
        a1[i]     = (_Float16)(sq[lrow * D_DIM + 32 + kb + i]      * qs);
        a1[i + 8] = (_Float16)(sq[lrow * D_DIM + 32 + kb + 8 + i]  * qs);
    }

    // ---- B stage copier: 64 codebook rows (8 KB) per round, 32 B/thread ----
    auto stageCopy = [&](int st, int bufi) {
        int row  = t >> 2;          // 0..63
        int part = t & 3;           // 4 x 16 halves per row
        const _Float16* g = cn_h + (size_t)(st * STAGE_COLS + row) * D_DIM + part * 16;
        _Float16* l = &sB[bufi][row * D_DIM + part * 16];
#if HVQ_ASYNC
        __builtin_amdgcn_global_load_async_to_lds_b128((gv4i*)g,       (lv4i*)l,       0, 0);
        __builtin_amdgcn_global_load_async_to_lds_b128((gv4i*)(g + 8), (lv4i*)(l + 8), 0, 0);
#else
        uint4 v0 = *(const uint4*)g;
        uint4 v1 = *(const uint4*)(g + 8);
        ((uint4*)l)[0] = v0;
        ((uint4*)l)[1] = v1;
#endif
    };

    const int nb  = L & 15;          // B column within 16-wide chunk
    const int kbb = (L >> 4) * 16;   // B lane-half K base

    float bestv[8];
    int   besti[8];
#pragma unroll
    for (int r = 0; r < 8; ++r) { bestv[r] = -INFINITY; besti[r] = 0; }

    stageCopy(0, 0);

    for (int st = 0; st < N_STAGES; ++st) {
#if HVQ_ASYNC
        hvq_wait_async0();
#endif
        __syncthreads();
        if (st + 1 < N_STAGES) stageCopy(st + 1, (st + 1) & 1);

        const _Float16* cur = sB[st & 1];
        const int s0 = st * STAGE_COLS;

#pragma unroll
        for (int cp = 0; cp < 2; ++cp) {
            const int cA = cp * 32;
            const int cB = cp * 32 + 16;
            const _Float16* pA = cur + (cA + nb) * D_DIM + kbb;
            const _Float16* pB = cur + (cB + nb) * D_DIM + kbb;
            v16h bA0 = *(const v16h*)pA;
            v16h bA1 = *(const v16h*)(pA + 32);
            v16h bB0 = *(const v16h*)pB;
            v16h bB1 = *(const v16h*)(pB + 32);

            v8f accA = {}, accB = {};
            accA = __builtin_amdgcn_wmma_f32_16x16x32_f16(false, a0, false, bA0,
                                                          (short)0, accA, false, false);
            accB = __builtin_amdgcn_wmma_f32_16x16x32_f16(false, a0, false, bB0,
                                                          (short)0, accB, false, false);
            accA = __builtin_amdgcn_wmma_f32_16x16x32_f16(false, a1, false, bA1,
                                                          (short)0, accA, false, false);
            accB = __builtin_amdgcn_wmma_f32_16x16x32_f16(false, a1, false, bB1,
                                                          (short)0, accB, false, false);

            const int colA = s0 + cA + nb;
            const int colB = s0 + cB + nb;
#pragma unroll
            for (int r = 0; r < 8; ++r) {
                if (accA[r] > bestv[r]) { bestv[r] = accA[r]; besti[r] = colA; }
            }
#pragma unroll
            for (int r = 0; r < 8; ++r) {
                if (accB[r] > bestv[r]) { bestv[r] = accB[r]; besti[r] = colB; }
            }
        }
    }

    // ---- argmax reduction across the 16 lanes of each half ----
#pragma unroll
    for (int msk = 1; msk <= 8; msk <<= 1) {
#pragma unroll
        for (int r = 0; r < 8; ++r) {
            float ov = __shfl_xor(bestv[r], msk, 32);
            int   oi = __shfl_xor(besti[r], msk, 32);
            if (ov > bestv[r] || (ov == bestv[r] && oi < besti[r])) {
                bestv[r] = ov; besti[r] = oi;
            }
        }
    }

    // ---- emit: idx, histogram atomics, gathered codebook rows ----
    const float* cbh = cb + (size_t)h * S_DIM * D_DIM;
    float* idx_h = idxp + ((size_t)(b * H_DIM + h)) * N_DIM;
#pragma unroll
    for (int r = 0; r < 8; ++r) {
        int ilo = __shfl(besti[r], 0, 32);
        int ihi = __shfl(besti[r], 16, 32);
        int nlo = n0 + w * 16 + r;
        int nhi = nlo + 8;
        if (L == 0) {
            idx_h[nlo] = (float)ilo;
            atomicAdd(&counts[h * S_DIM + ilo], 1.0f);
        }
        if (L == 16) {
            idx_h[nhi] = (float)ihi;
            atomicAdd(&counts[h * S_DIM + ihi], 1.0f);
        }
        const float2* slo = (const float2*)(cbh + (size_t)ilo * D_DIM) + L;
        const float2* shi = (const float2*)(cbh + (size_t)ihi * D_DIM) + L;
        float2* dlo = (float2*)(outp + ((size_t)b * N_DIM + nlo) * F_DIM + h * D_DIM) + L;
        float2* dhi = (float2*)(outp + ((size_t)b * N_DIM + nhi) * F_DIM + h * D_DIM) + L;
        *dlo = *slo;
        *dhi = *shi;
    }
}

// ---------------------------------------------------------------------------
// Kernel C: per-head perplexity from histogram.
// ---------------------------------------------------------------------------
__global__ void hvq_perp_kernel(const float* __restrict__ counts,
                                float* __restrict__ perp) {
    __shared__ float red[256];
    const int hh = blockIdx.x;
    const int t  = threadIdx.x;
    float local = 0.f;
#pragma unroll
    for (int j = 0; j < 4; ++j) {
        float cval = counts[hh * S_DIM + t + j * 256];
        float mean = cval * (1.0f / (float)(B_DIM * N_DIM));
        local += mean * logf(mean + 1e-10f);
    }
    red[t] = local;
    __syncthreads();
    for (int stride = 128; stride > 0; stride >>= 1) {
        if (t < stride) red[t] += red[t + stride];
        __syncthreads();
    }
    if (t == 0) perp[hh] = expf(-red[0]);
}

// ---------------------------------------------------------------------------
extern "C" void kernel_launch(void* const* d_in, const int* in_sizes, int n_in,
                              void* d_out, int out_size, void* d_ws, size_t ws_size,
                              hipStream_t stream) {
    (void)in_sizes; (void)n_in; (void)out_size; (void)ws_size;
    const float* x  = (const float*)d_in[0];   // [8, 4096, 512] f32
    const float* cb = (const float*)d_in[1];   // [8, 1024, 64]  f32

    float* outp  = (float*)d_out;
    float* idxp  = outp + (size_t)B_DIM * N_DIM * F_DIM;
    float* perpp = idxp + (size_t)B_DIM * H_DIM * N_DIM;

    float*    counts = (float*)d_ws;
    _Float16* cnh    = (_Float16*)((char*)d_ws + 32768);

    (void)hipMemsetAsync(counts, 0, H_DIM * S_DIM * sizeof(float), stream);

    hvq_cbnorm_kernel<<<(H_DIM * S_DIM + 255) / 256, 256, 0, stream>>>(cb, cnh);

    dim3 grid(B_DIM * H_DIM, N_DIM / ROWS_PER_BLOCK);
    hvq_main_kernel<<<grid, 256, 0, stream>>>(x, cb, cnh, outp, idxp, counts);

    hvq_perp_kernel<<<H_DIM, 256, 0, stream>>>(counts, perpp);
}
